// FoveaHead_69733089018390
// MI455X (gfx1250) — matrix operand order repaired and checked by reading, not compiled, not run on hardware
//
#include <hip/hip_runtime.h>
#include <hip/hip_bf16.h>

// ---------------------------------------------------------------------------
// FoveaHead on MI455X (gfx1250): implicit-GEMM 3x3 convs via v_wmma bf16.
// Compute-bound (~524 GFLOP vs ~0.1 GB HBM traffic at 23.3 TB/s). Conv inner
// loop: V_WMMA_F32_16X16X32_BF16, tap-major K (tap uniform per K-step), a
// coalesced im2col fill (pointer += 32*P per element), and a 256-Cout x 64-pos
// workgroup tile: 8 wmma per K-step per wave, A-weights reused x4 across N.
// ---------------------------------------------------------------------------

typedef __attribute__((ext_vector_type(16))) __bf16 v16bf;
typedef __attribute__((ext_vector_type(8)))  float  v8f;

#define NTOT 2843   // 1000 + 1000 + 625 + 169 + 49
#define BATCH 4
#define CH 256
#define PTOT 13343

// ---------------------------------------------------------------------------
// flat f32 -> bf16 convert (feats)
// ---------------------------------------------------------------------------
__global__ void cvt_f32_bf16_kernel(const float* __restrict__ in,
                                    __bf16* __restrict__ out, long long n) {
    long long i = (long long)blockIdx.x * 256 + threadIdx.x;
    if (i < n) out[i] = (__bf16)in[i];
}

// ---------------------------------------------------------------------------
// weight convert + reorder: OIHW [co][ci][kh][kw] f32 -> [co][t*256+ci] bf16
// (tap-major K so the conv's K-step has a uniform tap)
// ---------------------------------------------------------------------------
__global__ void cvt_w_kernel(const float* __restrict__ in,
                             __bf16* __restrict__ out, int CoTot) {
    long long idx = (long long)blockIdx.x * 256 + threadIdx.x;
    long long total = (long long)CoTot * 2304;
    if (idx >= total) return;
    int co  = (int)(idx / 2304);
    int rem = (int)(idx - (long long)co * 2304);
    int t   = rem >> 8;      // 0..8  (kh*3+kw)
    int ci  = rem & 255;
    out[idx] = (__bf16)in[((long long)co * 256 + ci) * 9 + t];
}

// ---------------------------------------------------------------------------
// Implicit-GEMM 3x3 SAME conv, bf16 in / f32 accum via WMMA.
//   xin : [B][256][H][W] bf16
//   wK  : [Cout][9*256] bf16, tap-major K
//   out : ((b*Cout + co) * outStride + p), bf16+ReLU (mid) or f32 (final)
// block = 256 threads = 8 waves. Workgroup tile: 256 Cout x 64 positions.
// Wave w: M-subtiles {w*16, w*16+128}, N-subtiles {0..3} -> 8 wmma/step.
// Fill: lane -> position (coalesced, two 32-position halves), (wave,i) -> kIdx;
// per-tap address math hoisted, per step it is a single += 32*P per element.
// ---------------------------------------------------------------------------
__global__ __launch_bounds__(256)
void conv3x3_wmma(const __bf16* __restrict__ xin,
                  const __bf16* __restrict__ wK,
                  const float*  __restrict__ bias,
                  __bf16* __restrict__ out_bf,
                  float*  __restrict__ out_f32,
                  int Cout, int H, int W,
                  int outStride, int relu)
{
    const int P   = H * W;
    const int K   = CH * 9;            // 2304
    const int b   = blockIdx.z;
    const int tid = threadIdx.x;
    const int wv  = tid >> 5;
    const int lane = tid & 31;
    const int half = lane >> 4;
    const int r    = lane & 15;
    const int n0   = blockIdx.x * 64;

    __shared__ __bf16 ldsB[64 * 32];   // [n][kIdx]

    const __bf16* xb = xin + (size_t)b * CH * P;

    // ---- fill-role constants: lane = position (two halves), (wave,i) = kIdx
    const int pA = n0 + lane;
    const int pB = n0 + 32 + lane;
    const bool vA = (pA < P);
    const bool vB = (pB < P);
    const int pcA = vA ? pA : 0;
    const int pcB = vB ? pB : 0;
    const int yA = pcA / W, xA = pcA - yA * W;
    const int yB = pcB / W, xB = pcB - yB * W;
    const int stride32P = 32 * P;
    __bf16* stA = &ldsB[lane * 32 + wv];          // +8,+16,+24 for i=1..3
    __bf16* stB = &ldsB[(lane + 32) * 32 + wv];

    // ---- A row pointers (2 M-subtiles), clamped for small Cout ----
    int row0 = wv * 16 + r;        if (row0 >= Cout) row0 = Cout - 1;
    int row1 = wv * 16 + 128 + r;  if (row1 >= Cout) row1 = Cout - 1;
    const __bf16* pw0 = wK + (size_t)row0 * K + 16 * half;
    const __bf16* pw1 = wK + (size_t)row1 * K + 16 * half;

    // ---- B fragment LDS read pointers (4 N-subtiles) ----
    const uint4* pbN[4];
#pragma unroll
    for (int ns = 0; ns < 4; ++ns)
        pbN[ns] = reinterpret_cast<const uint4*>(&ldsB[(ns * 16 + r) * 32 + 16 * half]);

    v8f acc00 = {}, acc01 = {}, acc02 = {}, acc03 = {};
    v8f acc10 = {}, acc11 = {}, acc12 = {}, acc13 = {};
    const __bf16 z = (__bf16)0.0f;

    for (int t = 0; t < 9; ++t) {
        // per-tap (uniform per 8 K-steps) address setup
        const int kh = t / 3;
        const int kw = t - kh * 3;
        const int iyA = yA + kh - 1, ixA = xA + kw - 1;
        const int iyB = yB + kh - 1, ixB = xB + kw - 1;
        const bool okA = vA && ((unsigned)iyA < (unsigned)H) && ((unsigned)ixA < (unsigned)W);
        const bool okB = vB && ((unsigned)iyB < (unsigned)H) && ((unsigned)ixB < (unsigned)W);
        const int baseA = iyA * W + ixA;
        const int baseB = iyB * W + ixB;
        int oA0 = (wv     ) * P + baseA, oA1 = (wv +  8) * P + baseA;
        int oA2 = (wv + 16) * P + baseA, oA3 = (wv + 24) * P + baseA;
        int oB0 = (wv     ) * P + baseB, oB1 = (wv +  8) * P + baseB;
        int oB2 = (wv + 16) * P + baseB, oB3 = (wv + 24) * P + baseB;
        __bf16 gA0 = okA ? xb[oA0] : z, gA1 = okA ? xb[oA1] : z;
        __bf16 gA2 = okA ? xb[oA2] : z, gA3 = okA ? xb[oA3] : z;
        __bf16 gB0 = okB ? xb[oB0] : z, gB1 = okB ? xb[oB1] : z;
        __bf16 gB2 = okB ? xb[oB2] : z, gB3 = okB ? xb[oB3] : z;

#pragma unroll
        for (int s8 = 0; s8 < 8; ++s8) {
            stA[0]  = gA0; stA[8]  = gA1; stA[16] = gA2; stA[24] = gA3;
            stB[0]  = gB0; stB[8]  = gB1; stB[16] = gB2; stB[24] = gB3;
            __syncthreads();

            // prefetch next K-step's fill (overlaps with wmma below)
            if (s8 < 7) {
                oA0 += stride32P; oA1 += stride32P; oA2 += stride32P; oA3 += stride32P;
                oB0 += stride32P; oB1 += stride32P; oB2 += stride32P; oB3 += stride32P;
                gA0 = okA ? xb[oA0] : z; gA1 = okA ? xb[oA1] : z;
                gA2 = okA ? xb[oA2] : z; gA3 = okA ? xb[oA3] : z;
                gB0 = okB ? xb[oB0] : z; gB1 = okB ? xb[oB1] : z;
                gB2 = okB ? xb[oB2] : z; gB3 = okB ? xb[oB3] : z;
            }

            union { uint4 u[2]; v16bf v; } a0, a1, b0, b1, b2, b3;
            a0.u[0] = *reinterpret_cast<const uint4*>(pw0);
            a0.u[1] = *reinterpret_cast<const uint4*>(pw0 + 8);
            a1.u[0] = *reinterpret_cast<const uint4*>(pw1);
            a1.u[1] = *reinterpret_cast<const uint4*>(pw1 + 8);
            pw0 += 32; pw1 += 32;
            b0.u[0] = pbN[0][0]; b0.u[1] = pbN[0][1];
            b1.u[0] = pbN[1][0]; b1.u[1] = pbN[1][1];
            b2.u[0] = pbN[2][0]; b2.u[1] = pbN[2][1];
            b3.u[0] = pbN[3][0]; b3.u[1] = pbN[3][1];

            acc00 = __builtin_amdgcn_wmma_f32_16x16x32_bf16(false, a0.v, false, b0.v, (short)0, acc00, false, false);
            acc01 = __builtin_amdgcn_wmma_f32_16x16x32_bf16(false, a0.v, false, b1.v, (short)0, acc01, false, false);
            acc02 = __builtin_amdgcn_wmma_f32_16x16x32_bf16(false, a0.v, false, b2.v, (short)0, acc02, false, false);
            acc03 = __builtin_amdgcn_wmma_f32_16x16x32_bf16(false, a0.v, false, b3.v, (short)0, acc03, false, false);
            acc10 = __builtin_amdgcn_wmma_f32_16x16x32_bf16(false, a1.v, false, b0.v, (short)0, acc10, false, false);
            acc11 = __builtin_amdgcn_wmma_f32_16x16x32_bf16(false, a1.v, false, b1.v, (short)0, acc11, false, false);
            acc12 = __builtin_amdgcn_wmma_f32_16x16x32_bf16(false, a1.v, false, b2.v, (short)0, acc12, false, false);
            acc13 = __builtin_amdgcn_wmma_f32_16x16x32_bf16(false, a1.v, false, b3.v, (short)0, acc13, false, false);
            __syncthreads();
        }
    }

    // ---- epilogue: C/D layout: vgpr j, lanes 0-15 -> M=j, lanes 16-31 -> M=8+j
    const int pN[4] = { n0 + r, n0 + 16 + r, n0 + 32 + r, n0 + 48 + r };
#pragma unroll
    for (int j = 0; j < 8; ++j) {
        const int m   = j + 8 * half;
        const int coA = wv * 16 + m;
        const int coB = wv * 16 + 128 + m;
        float va[4] = { acc00[j], acc01[j], acc02[j], acc03[j] };
        float vb[4] = { acc10[j], acc11[j], acc12[j], acc13[j] };
        if (coA < Cout) {
            float bia = bias[coA];
            size_t rowOff = ((size_t)b * Cout + coA) * (size_t)outStride;
#pragma unroll
            for (int ns = 0; ns < 4; ++ns) {
                float v = va[ns] + bia;
                if (relu) v = v > 0.f ? v : 0.f;
                if (pN[ns] < P) {
                    if (out_f32) out_f32[rowOff + pN[ns]] = v;
                    else         out_bf[rowOff + pN[ns]]  = (__bf16)v;
                }
            }
        }
        if (coB < Cout) {
            float bib = bias[coB];
            size_t rowOff = ((size_t)b * Cout + coB) * (size_t)outStride;
#pragma unroll
            for (int ns = 0; ns < 4; ++ns) {
                float v = vb[ns] + bib;
                if (relu) v = v > 0.f ? v : 0.f;
                if (pN[ns] < P) {
                    if (out_f32) out_f32[rowOff + pN[ns]] = v;
                    else         out_bf[rowOff + pN[ns]]  = (__bf16)v;
                }
            }
        }
    }
}

// ---------------------------------------------------------------------------
// decode one level: per-position class max/argmax + exp box decode
// ---------------------------------------------------------------------------
__global__ void decode_kernel(const float* __restrict__ cp,   // [B][80][PTOT]
                              const float* __restrict__ bp,   // [B][4][PTOT]
                              float* __restrict__ boxes_all,  // [B][PTOT][4]
                              float* __restrict__ score_all,  // [B][PTOT]
                              int*   __restrict__ cls_all,    // [B][PTOT]
                              int Poff, int P, int W, float s, float rl)
{
    int b = blockIdx.y;
    int p = blockIdx.x * blockDim.x + threadIdx.x;
    if (p >= P) return;
    const float* cpb = cp + (size_t)b * 80 * PTOT + Poff + p;
    float best = cpb[0];
    int   bc   = 0;
    for (int c = 1; c < 80; ++c) {
        float v = cpb[(size_t)c * PTOT];
        if (v > best) { best = v; bc = c; }
    }
    int g = Poff + p;
    score_all[(size_t)b * PTOT + g] = best;
    cls_all[(size_t)b * PTOT + g]   = bc;

    const float* bpb = bp + (size_t)b * 4 * PTOT + Poff + p;
    float e0 = __expf(bpb[0]);
    float e1 = __expf(bpb[(size_t)PTOT]);
    float e2 = __expf(bpb[2 * (size_t)PTOT]);
    float e3 = __expf(bpb[3 * (size_t)PTOT]);
    int y = p / W, x = p - y * W;
    float cx = s * (x + 0.5f), cy = s * (y + 0.5f);
    float* bo = boxes_all + ((size_t)b * PTOT + g) * 4;
    bo[0] = cx - rl * e0;
    bo[1] = cy - rl * e1;
    bo[2] = cx + rl * e2;
    bo[3] = cy + rl * e3;
}

// ---------------------------------------------------------------------------
// per (batch) block: top-k selection (stable descending) or natural-order copy
// when P <= k (matches the reference's branch).
// ---------------------------------------------------------------------------
__global__ __launch_bounds__(256)
void select_kernel(const float* __restrict__ score_all, const int* __restrict__ cls_all,
                   const float* __restrict__ boxes_all, float* __restrict__ score_tmp,
                   float* __restrict__ scores_sel, int* __restrict__ labels_sel,
                   float* __restrict__ boxes_sel,
                   int Poff, int P, int selOff, int Ksel)
{
    int b = blockIdx.x, tid = threadIdx.x;
    const float* sa = score_all + (size_t)b * PTOT + Poff;
    const int*   ca = cls_all   + (size_t)b * PTOT + Poff;
    const float* ba = boxes_all + ((size_t)b * PTOT + Poff) * 4;
    float* so = scores_sel + (size_t)b * NTOT + selOff;
    int*   lo = labels_sel + (size_t)b * NTOT + selOff;
    float* bo = boxes_sel  + ((size_t)b * NTOT + selOff) * 4;

    if (P <= Ksel) {
        for (int i = tid; i < P; i += 256) {
            so[i] = sa[i];
            lo[i] = ca[i];
            bo[i * 4 + 0] = ba[i * 4 + 0];
            bo[i * 4 + 1] = ba[i * 4 + 1];
            bo[i * 4 + 2] = ba[i * 4 + 2];
            bo[i * 4 + 3] = ba[i * 4 + 3];
        }
        return;
    }

    float* st = score_tmp + (size_t)b * PTOT + Poff;
    for (int i = tid; i < P; i += 256) st[i] = sa[i];
    __syncthreads();

    __shared__ float rv[256];
    __shared__ int   ri[256];
    for (int k = 0; k < Ksel; ++k) {
        float bv = -3.4e38f;
        int   bi = 0;
        for (int i = tid; i < P; i += 256) {
            float v = st[i];
            if (v > bv) { bv = v; bi = i; }
        }
        rv[tid] = bv; ri[tid] = bi;
        __syncthreads();
        for (int o = 128; o > 0; o >>= 1) {
            if (tid < o) {
                if (rv[tid + o] > rv[tid] ||
                    (rv[tid + o] == rv[tid] && ri[tid + o] < ri[tid])) {
                    rv[tid] = rv[tid + o];
                    ri[tid] = ri[tid + o];
                }
            }
            __syncthreads();
        }
        if (tid == 0) {
            int wi = ri[0];
            so[k] = rv[0];
            lo[k] = ca[wi];
            bo[k * 4 + 0] = ba[wi * 4 + 0];
            bo[k * 4 + 1] = ba[wi * 4 + 1];
            bo[k * 4 + 2] = ba[wi * 4 + 2];
            bo[k * 4 + 3] = ba[wi * 4 + 3];
            st[wi] = -3.4e38f;
        }
        __syncthreads();
    }
}

// ---------------------------------------------------------------------------
// batched NMS (per-class coordinate-offset trick) mirroring the lax.scan
// semantics (kept box also suppresses itself afterwards; recorded first).
// ---------------------------------------------------------------------------
__global__ __launch_bounds__(256)
void nms_kernel(const float* __restrict__ boxes_sel, const float* __restrict__ scores_sel,
                const int* __restrict__ labels_sel, float* __restrict__ nmsBoxes,
                int* __restrict__ order, float* __restrict__ keep_out)
{
    int b = blockIdx.x, tid = threadIdx.x;
    const float* bb = boxes_sel  + (size_t)b * NTOT * 4;
    const float* sc = scores_sel + (size_t)b * NTOT;
    const int*   lb = labels_sel + (size_t)b * NTOT;
    float* nb = nmsBoxes + (size_t)b * NTOT * 4;
    int*   od = order    + (size_t)b * NTOT;

    __shared__ float red[256];
    float mx = -3.4e38f;
    for (int i = tid; i < NTOT * 4; i += 256) mx = fmaxf(mx, bb[i]);
    red[tid] = mx;
    __syncthreads();
    for (int o = 128; o > 0; o >>= 1) {
        if (tid < o) red[tid] = fmaxf(red[tid], red[tid + o]);
        __syncthreads();
    }
    float maxc = red[0] + 1.0f;
    __syncthreads();

    for (int i = tid; i < NTOT; i += 256) {
        float offv = (float)lb[i] * maxc;
        nb[i * 4 + 0] = bb[i * 4 + 0] + offv;
        nb[i * 4 + 1] = bb[i * 4 + 1] + offv;
        nb[i * 4 + 2] = bb[i * 4 + 2] + offv;
        nb[i * 4 + 3] = bb[i * 4 + 3] + offv;
    }

    // stable descending argsort via rank counting
    for (int i = tid; i < NTOT; i += 256) {
        float si = sc[i];
        int rnk = 0;
        for (int j = 0; j < NTOT; ++j) {
            float sj = sc[j];
            rnk += (sj > si) || (sj == si && j < i);
        }
        od[rnk] = i;
    }
    __threadfence_block();
    __syncthreads();

    __shared__ unsigned char sup[NTOT];
    __shared__ unsigned char keepA[NTOT];
    for (int i = tid; i < NTOT; i += 256) { sup[i] = 0; keepA[i] = 0; }
    __syncthreads();

    for (int cpos = 0; cpos < NTOT; ++cpos) {
        int c  = od[cpos];
        int kc = sup[c] ? 0 : 1;
        __syncthreads();
        if (kc) {
            float ax1 = nb[c * 4 + 0], ay1 = nb[c * 4 + 1];
            float ax2 = nb[c * 4 + 2], ay2 = nb[c * 4 + 3];
            float areaA = (ax2 - ax1) * (ay2 - ay1);
            for (int j = tid; j < NTOT; j += 256) {
                float bx1 = nb[j * 4 + 0], by1 = nb[j * 4 + 1];
                float bx2 = nb[j * 4 + 2], by2 = nb[j * 4 + 3];
                float areaB = (bx2 - bx1) * (by2 - by1);
                float lx = fmaxf(ax1, bx1), ly = fmaxf(ay1, by1);
                float rx = fminf(ax2, bx2), ry = fminf(ay2, by2);
                float w = fmaxf(rx - lx, 0.0f), h = fmaxf(ry - ly, 0.0f);
                float inter = w * h;
                float uni = fmaxf(areaA + areaB - inter, 1e-9f);
                if (inter / uni > 0.5f) sup[j] = 1;
            }
        }
        if (tid == 0) keepA[c] = (unsigned char)kc;
        __syncthreads();
    }
    for (int i = tid; i < NTOT; i += 256)
        keep_out[(size_t)b * NTOT + i] = keepA[i] ? 1.0f : 0.0f;
}

// ---------------------------------------------------------------------------
// pack boxes / scores / labels+1 into d_out (keep written by nms_kernel)
// ---------------------------------------------------------------------------
__global__ void writeout_kernel(const float* __restrict__ boxes_sel,
                                const float* __restrict__ scores_sel,
                                const int* __restrict__ labels_sel,
                                float* __restrict__ out)
{
    const int BT = BATCH * NTOT;
    int i = blockIdx.x * 256 + threadIdx.x;
    if (i >= BT) return;
    out[i * 4 + 0] = boxes_sel[i * 4 + 0];
    out[i * 4 + 1] = boxes_sel[i * 4 + 1];
    out[i * 4 + 2] = boxes_sel[i * 4 + 2];
    out[i * 4 + 3] = boxes_sel[i * 4 + 3];
    out[(size_t)BT * 4 + i] = scores_sel[i];
    out[(size_t)BT * 5 + i] = (float)(labels_sel[i] + 1);
}

// ---------------------------------------------------------------------------
extern "C" void kernel_launch(void* const* d_in, const int* in_sizes, int n_in,
                              void* d_out, int out_size, void* d_ws, size_t ws_size,
                              hipStream_t stream) {
    (void)in_sizes; (void)n_in; (void)out_size; (void)ws_size;

    const float* feat[5] = {(const float*)d_in[0], (const float*)d_in[1],
                            (const float*)d_in[2], (const float*)d_in[3],
                            (const float*)d_in[4]};
    const float* cls_wmid = (const float*)d_in[5];
    const float* cls_bmid = (const float*)d_in[6];
    const float* cls_wout = (const float*)d_in[7];
    const float* cls_bout = (const float*)d_in[8];
    const float* box_wmid = (const float*)d_in[9];
    const float* box_bmid = (const float*)d_in[10];
    const float* box_wout = (const float*)d_in[11];
    const float* box_bout = (const float*)d_in[12];
    // d_in[13] = image_sizes: unused (reference's clamp is a no-op)

    static const int   Hs[5]     = {100, 50, 25, 13, 7};
    static const int   Ws_[5]    = {100, 50, 25, 13, 7};
    static const int   Ps[5]     = {10000, 2500, 625, 169, 49};
    static const int   Poff[5]   = {0, 10000, 12500, 13125, 13294};
    static const int   selOff[5] = {0, 1000, 2000, 2625, 2794};
    static const float Ss[5]     = {8.f, 16.f, 32.f, 64.f, 128.f};
    static const float Rl[5]     = {16.f, 32.f, 64.f, 128.f, 256.f};
    const int Kw = CH * 9;   // 2304

    // ---- workspace bump allocator (~100 MB total) ----
    char* ws = (char*)d_ws;
    size_t off = 0;
    auto alloc = [&](size_t bytes) -> void* {
        off = (off + 255) & ~(size_t)255;
        void* p = ws + off;
        off += bytes;
        return p;
    };
    __bf16* featsBf  = (__bf16*)alloc((size_t)BATCH * CH * PTOT * 2);
    __bf16* actA     = (__bf16*)alloc((size_t)BATCH * CH * 10000 * 2);
    __bf16* actB     = (__bf16*)alloc((size_t)BATCH * CH * 10000 * 2);
    __bf16* wClsMid  = (__bf16*)alloc((size_t)4 * CH * Kw * 2);
    __bf16* wClsOut  = (__bf16*)alloc((size_t)80 * Kw * 2);
    __bf16* wBoxMid  = (__bf16*)alloc((size_t)4 * CH * Kw * 2);
    __bf16* wBoxOut  = (__bf16*)alloc((size_t)4 * Kw * 2);
    float*  cpv      = (float*)alloc((size_t)BATCH * 80 * PTOT * 4);
    float*  bpv      = (float*)alloc((size_t)BATCH * 4 * PTOT * 4);
    float*  boxesAll = (float*)alloc((size_t)BATCH * PTOT * 4 * 4);
    float*  scoreAll = (float*)alloc((size_t)BATCH * PTOT * 4);
    int*    clsAll   = (int*)alloc((size_t)BATCH * PTOT * 4);
    float*  scoreTmp = (float*)alloc((size_t)BATCH * PTOT * 4);
    float*  boxesSel = (float*)alloc((size_t)BATCH * NTOT * 4 * 4);
    float*  scoresSel= (float*)alloc((size_t)BATCH * NTOT * 4);
    int*    labelsSel= (int*)alloc((size_t)BATCH * NTOT * 4);
    float*  nmsBoxes = (float*)alloc((size_t)BATCH * NTOT * 4 * 4);
    int*    order    = (int*)alloc((size_t)BATCH * NTOT * 4);

    // ---- convert feats (flat) + weights (tap-major reorder) to bf16 ----
    size_t featOff[5];
    {
        size_t fo = 0;
        for (int l = 0; l < 5; ++l) { featOff[l] = fo; fo += (size_t)BATCH * CH * Ps[l]; }
    }
    for (int l = 0; l < 5; ++l) {
        long long n = (long long)BATCH * CH * Ps[l];
        cvt_f32_bf16_kernel<<<(int)((n + 255) / 256), 256, 0, stream>>>(
            feat[l], featsBf + featOff[l], n);
    }
    auto cvtw = [&](const float* src, __bf16* dst, int CoTot) {
        long long n = (long long)CoTot * Kw;
        cvt_w_kernel<<<(int)((n + 255) / 256), 256, 0, stream>>>(src, dst, CoTot);
    };
    cvtw(cls_wmid, wClsMid, 4 * CH);
    cvtw(cls_wout, wClsOut, 80);
    cvtw(box_wmid, wBoxMid, 4 * CH);
    cvtw(box_wout, wBoxOut, 4);

    // ---- conv heads (5 levels x 2 heads x (4 mid + 1 out)) ----
    for (int l = 0; l < 5; ++l) {
        int H = Hs[l], W = Ws_[l], P = Ps[l];
        dim3 gridConv((P + 63) / 64, 1, BATCH);
        for (int head = 0; head < 2; ++head) {
            const __bf16* wm = head ? wBoxMid : wClsMid;
            const float*  bm = head ? box_bmid : cls_bmid;
            const __bf16* wo = head ? wBoxOut : wClsOut;
            const float*  bo = head ? box_bout : cls_bout;
            int   CoutO = head ? 4 : 80;
            float* outp = (head ? bpv : cpv) + Poff[l];

            const __bf16* src = featsBf + featOff[l];
            __bf16* dst = actA;
            for (int layer = 0; layer < 4; ++layer) {
                conv3x3_wmma<<<gridConv, 256, 0, stream>>>(
                    src, wm + (size_t)layer * CH * Kw, bm + layer * CH,
                    dst, nullptr, CH, H, W, P, 1);
                src = dst;
                dst = (dst == actA) ? actB : actA;
            }
            conv3x3_wmma<<<gridConv, 256, 0, stream>>>(
                src, wo, bo, nullptr, outp, CoutO, H, W, PTOT, 0);
        }
    }

    // ---- decode each level ----
    for (int l = 0; l < 5; ++l) {
        dim3 grid((Ps[l] + 255) / 256, BATCH);
        decode_kernel<<<grid, 256, 0, stream>>>(cpv, bpv, boxesAll, scoreAll, clsAll,
                                                Poff[l], Ps[l], Ws_[l], Ss[l], Rl[l]);
    }

    // ---- top-k / copy selection ----
    for (int l = 0; l < 5; ++l) {
        int Ksel = Ps[l] < 1000 ? Ps[l] : 1000;
        select_kernel<<<BATCH, 256, 0, stream>>>(scoreAll, clsAll, boxesAll, scoreTmp,
                                                 scoresSel, labelsSel, boxesSel,
                                                 Poff[l], Ps[l], selOff[l], Ksel);
    }

    // ---- NMS + output packing ----
    float* outF = (float*)d_out;
    nms_kernel<<<BATCH, 256, 0, stream>>>(boxesSel, scoresSel, labelsSel,
                                          nmsBoxes, order,
                                          outF + (size_t)BATCH * NTOT * 6);
    writeout_kernel<<<(BATCH * NTOT + 255) / 256, 256, 0, stream>>>(
        boxesSel, scoresSel, labelsSel, outF);
}
